// MaskedSelfAttention_46351287058709
// MI455X (gfx1250) — matrix operand order repaired and checked by reading, not compiled, not run on hardware
//
#include <hip/hip_runtime.h>

// ---------------------------------------------------------------------------
// Fused causal MHA for MI455X (gfx1250, wave32, WMMA).
// - All GEMM-shaped math through v_wmma_f32_16x16x32_bf16 (bf16 in, f32 acc).
// - Attention K/V tiles staged to LDS via GLOBAL_LOAD_ASYNC_TO_LDS_B128
//   (ASYNCcnt) when the builtin is available; plain 16B copies otherwise.
// - Intermediates (rope'd Q/K, V, context) kept bf16 in workspace.
// ---------------------------------------------------------------------------

typedef __bf16 bf16;
typedef __attribute__((ext_vector_type(16))) __bf16 v16bf;
typedef __attribute__((ext_vector_type(8)))  float  v8f;
typedef int v4i __attribute__((ext_vector_type(4)));

#define WMMA_BF16(A, Bm, C) \
  __builtin_amdgcn_wmma_f32_16x16x32_bf16(false, (A), false, (Bm), (short)0, (C), false, false)

constexpr int Bb = 2, Ss = 2048, Ee = 1024, Hh = 16, Dk = 64, HD = 1024;

__device__ inline v8f vzero8() {
  v8f z;
#pragma unroll
  for (int i = 0; i < 8; ++i) z[i] = 0.f;
  return z;
}

// ---- async global->LDS copy (gfx1250 ASYNCcnt path), safe fallback --------
#if defined(__has_builtin)
#if __has_builtin(__builtin_amdgcn_global_load_async_to_lds_b128)
#define HAVE_ASYNC_LDS 1
#endif
#endif

typedef v4i __attribute__((address_space(1)))* gv4i_ptr;   // global 128-bit
typedef v4i __attribute__((address_space(3)))* lv4i_ptr;   // LDS    128-bit

__device__ inline void async_copy16(bf16* lds_dst, const bf16* gsrc) {
#ifdef HAVE_ASYNC_LDS
  __builtin_amdgcn_global_load_async_to_lds_b128(
      (gv4i_ptr)gsrc, (lv4i_ptr)lds_dst, 0, 0);
#else
  *(uint4*)lds_dst = *(const uint4*)gsrc;
#endif
}

__device__ inline void wait_async_lds() {
#ifdef HAVE_ASYNC_LDS
#if __has_builtin(__builtin_amdgcn_s_wait_asynccnt)
  __builtin_amdgcn_s_wait_asynccnt(0);
#else
  asm volatile("s_wait_asynccnt 0" ::: "memory");
#endif
#endif
}

// ---------------------------------------------------------------------------
// Kernel 1: fused QKV projection + bias + RoPE, fp32 -> bf16.
// Grid: (B*S/64, 48).  nt<16 -> Q head nt, nt<32 -> K head nt-16, else V.
// 64(M) x 64(N) tile, K-loop over E in steps of 32; 4 waves, each owning a
// 16-row strip x 64 cols (4 wmma accumulators).
// ---------------------------------------------------------------------------
__global__ __launch_bounds__(128)
void qkv_rope_kernel(const float* __restrict__ x,
                     const float* __restrict__ wq, const float* __restrict__ bq,
                     const float* __restrict__ wk, const float* __restrict__ bk,
                     const float* __restrict__ wv, const float* __restrict__ bv,
                     bf16* __restrict__ Qo, bf16* __restrict__ Ko, bf16* __restrict__ Vo)
{
  __shared__ bf16 At[64][32];   // x rows  x K-chunk (bf16-converted)
  __shared__ bf16 Bt[64][32];   // weight rows (N features) x K-chunk

  const int t    = threadIdx.x;
  const int lane = t & 31;
  const int wid  = t >> 5;

  const int mt   = blockIdx.x;          // 64-row tile of flattened (B*S)
  const int nt   = blockIdx.y;          // 0..47
  const int sel  = nt >> 4;             // 0=q 1=k 2=v
  const int head = nt & 15;
  const int n0   = head * 64;

  const float* w    = (sel == 0) ? wq : (sel == 1) ? wk : wv;
  const float* bias = (sel == 0) ? bq : (sel == 1) ? bk : bv;
  bf16*        dst  = (sel == 0) ? Qo : (sel == 1) ? Ko : Vo;

  const int row0 = mt * 64;

  v8f acc[4];
#pragma unroll
  for (int n = 0; n < 4; ++n) acc[n] = vzero8();

  const int am  = wid * 16 + (lane & 15);   // A fragment row
  const int khi = (lane >> 4) * 8;          // A fragment K sub-offset
  const int kbB = (lane >> 4) * 16;         // B fragment K sub-offset

  for (int kk = 0; kk < Ee; kk += 32) {
    for (int i = t; i < 64 * 32; i += 128) {
      const int r = i >> 5, c = i & 31;
      At[r][c] = (bf16)x[(size_t)(row0 + r) * Ee + kk + c];
      Bt[r][c] = (bf16)w[(size_t)(n0 + r) * Ee + kk + c];
    }
    __builtin_prefetch(&w[(size_t)n0 * Ee + kk + 32], 0, 1);
    __syncthreads();

    v16bf a;
#pragma unroll
    for (int i = 0; i < 16; ++i) {
      const int k = (i < 8) ? (khi + i) : (16 + khi + (i - 8));
      a[i] = At[am][k];
    }
#pragma unroll
    for (int n = 0; n < 4; ++n) {
      const int bn = n * 16 + (lane & 15);
      v16bf b;
#pragma unroll
      for (int i = 0; i < 16; ++i) b[i] = Bt[bn][(i < 8) ? (kbB / 2 + i) : (16 + kbB / 2 + (i - 8))];
      acc[n] = WMMA_BF16(a, b, acc[n]);
    }
    __syncthreads();
  }

  // Epilogue: bias, RoPE (q,k), bf16 store head-major [b][h][s][d].
  const int mgrp = (lane >> 4) * 8;
  const float kLn10000 = 9.210340371976184f;   // ln(10000)
#pragma unroll
  for (int n = 0; n < 4; ++n) {
    const int d    = n * 16 + (lane & 15);
    const float bv_ = bias[n0 + d];
    // 1/theta_d = 10000^(-2*(d&31)/64) = exp(-(2*(d&31)/64) * ln(10000))
    const float invt = __expf((float)(d & 31) * (-2.f / 64.f) * kLn10000);
#pragma unroll
    for (int r = 0; r < 8; ++r) {
      const int row  = row0 + wid * 16 + mgrp + r;
      const int s    = row & (Ss - 1);
      const int bidx = row >> 11;             // row / S
      float v = acc[n][r] + bv_;
      if (sel < 2) {
        // pairs (2i,2i+1) sit in adjacent lanes of the C layout
        const float partner = __shfl_xor(v, 1, 32);
        const float rot     = (d & 1) ? partner : -partner;
        const float freq    = (float)s * invt;
        float sn, cs;
        __sincosf(freq, &sn, &cs);
        v = v * cs + rot * sn;
      }
      dst[(((size_t)bidx * Hh + head) * Ss + s) * 64 + d] = (bf16)v;
    }
  }
}

// ---------------------------------------------------------------------------
// Kernel 2: flash-style causal attention per (b,h).
// Grid: (S/64, B*H). 4 waves; wave w owns q rows qt*64+w*16 .. +15.
// Per 32-key step: K/V tiles (32x64 bf16 each) staged to LDS with async
// global->LDS copies (one 16B transfer per lane per tensor chunk), then
// S = Q K^T via 2x2 WMMA, online softmax (16-lane shuffle reductions),
// P staged through per-wave LDS (C->A layout), O += P V (4 WMMA).
// ---------------------------------------------------------------------------
__global__ __launch_bounds__(128)
void attention_kernel(const bf16* __restrict__ Q, const bf16* __restrict__ K,
                      const bf16* __restrict__ V, bf16* __restrict__ Ctx)
{
  __shared__ bf16 Kt[32][64];             // key tile  (32 keys x 64 d)
  __shared__ bf16 Vt[32][64];             // value tile
  __shared__ bf16 Pl[4][16][32];          // per-wave P tile (q x 32 keys)

  const int t    = threadIdx.x;
  const int lane = t & 31;
  const int wid  = t >> 5;

  const int qt   = blockIdx.x;            // 64-row q tile
  const int bh   = blockIdx.y;            // b*H + h
  const int bidx = bh >> 4, head = bh & 15;

  const size_t base = (size_t)bh * Ss * 64;
  const int m0   = qt * 64 + wid * 16;
  const int khi  = (lane >> 4) * 8;
  const int nlo  = lane & 15;
  const int mgrp = (lane >> 4) * 8;

  // Q fragments (two K=32 chunks over Dk=64), held in registers for whole loop
  const bf16* Qp = Q + base + (size_t)(m0 + nlo) * 64;
  v16bf qa[2];
#pragma unroll
  for (int c = 0; c < 2; ++c)
#pragma unroll
    for (int i = 0; i < 16; ++i) {
      const int d = c * 32 + ((i < 8) ? (khi + i) : (16 + khi + (i - 8)));
      qa[c][i] = Qp[d];
    }

  float mrow[8], lrow[8];
#pragma unroll
  for (int r = 0; r < 8; ++r) { mrow[r] = -1e30f; lrow[r] = 0.f; }
  v8f oacc[4];
#pragma unroll
  for (int n = 0; n < 4; ++n) oacc[n] = vzero8();

  const int nkb = 2 * qt + 2;             // causal bound: keys <= qt*64+63
  for (int kb = 0; kb < nkb; ++kb) {
    const int k0 = kb * 32;

    // ---- stage K/V tile into LDS (async, 16B per transfer) ----
#pragma unroll
    for (int j = 0; j < 2; ++j) {
      const int c   = t + j * 128;        // 0..255 chunk id (16B chunks)
      const int r   = c >> 3;             // key row 0..31
      const int col = (c & 7) * 8;        // d offset 0..56
      async_copy16(&Kt[r][col], K + base + (size_t)(k0 + r) * 64 + col);
      async_copy16(&Vt[r][col], V + base + (size_t)(k0 + r) * 64 + col);
    }
    __builtin_prefetch(K + base + (size_t)(k0 + 32 + (t >> 3)) * 64, 0, 1);
    wait_async_lds();
    __syncthreads();

    // ---- scores for 32 keys (two 16-key halves) ----
    v8f sc[2];
#pragma unroll
    for (int h2 = 0; h2 < 2; ++h2) {
      v8f s = vzero8();
#pragma unroll
      for (int c = 0; c < 2; ++c) {
        v16bf kf;
#pragma unroll
        for (int i = 0; i < 16; ++i)
          kf[i] = Kt[h2 * 16 + nlo][c * 32 + ((i < 8) ? (khi + i) : (16 + khi + (i - 8)))];
        s = WMMA_BF16(qa[c], kf, s);
      }
      sc[h2] = s;
    }

    // ---- online softmax update ----
#pragma unroll
    for (int r = 0; r < 8; ++r) {
      const int qi = m0 + mgrp + r;
      float s0 = (k0 + nlo      <= qi) ? sc[0][r] * 0.125f : -1e30f;
      float s1 = (k0 + 16 + nlo <= qi) ? sc[1][r] * 0.125f : -1e30f;
      float mx = fmaxf(s0, s1);
#pragma unroll
      for (int off = 1; off < 16; off <<= 1)
        mx = fmaxf(mx, __shfl_xor(mx, off, 32));
      const float mnew = fmaxf(mrow[r], mx);
      const float p0 = __expf(s0 - mnew);
      const float p1 = __expf(s1 - mnew);
      float rs = p0 + p1;
#pragma unroll
      for (int off = 1; off < 16; off <<= 1)
        rs += __shfl_xor(rs, off, 32);
      const float alpha = __expf(mrow[r] - mnew);
      mrow[r] = mnew;
      lrow[r] = lrow[r] * alpha + rs;
#pragma unroll
      for (int n = 0; n < 4; ++n) oacc[n][r] *= alpha;
      Pl[wid][mgrp + r][nlo]      = (bf16)p0;
      Pl[wid][mgrp + r][16 + nlo] = (bf16)p1;
    }

    // ---- O += P(16x32) * V(32x64) ----
    v16bf pa;
#pragma unroll
    for (int i = 0; i < 16; ++i) {
      const int k = (i < 8) ? (khi + i) : (16 + khi + (i - 8));
      pa[i] = Pl[wid][lane & 15][k];
    }
#pragma unroll
    for (int n = 0; n < 4; ++n) {
      v16bf vb;
#pragma unroll
      for (int i = 0; i < 16; ++i)
        vb[i] = Vt[(i < 8) ? (khi + i) : (16 + khi + (i - 8))][n * 16 + nlo];
      oacc[n] = WMMA_BF16(pa, vb, oacc[n]);
    }

    __syncthreads();   // all waves done with Kt/Vt before next stage
  }

  // Epilogue: normalize, store context [b][s][h*64+d] (bf16)
#pragma unroll
  for (int n = 0; n < 4; ++n) {
    const int d = n * 16 + nlo;
#pragma unroll
    for (int r = 0; r < 8; ++r) {
      const int s = m0 + mgrp + r;
      const float val = oacc[n][r] / lrow[r];
      Ctx[((size_t)bidx * Ss + s) * HD + head * 64 + d] = (bf16)val;
    }
  }
}

// ---------------------------------------------------------------------------
// Kernel 3: output projection out = ctx @ wo^T + bo (fp32 out).
// Grid: (B*S/64, E/64); same tiling as kernel 1.
// ---------------------------------------------------------------------------
__global__ __launch_bounds__(128)
void out_proj_kernel(const bf16* __restrict__ Ctx, const float* __restrict__ wo,
                     const float* __restrict__ bo, float* __restrict__ out)
{
  __shared__ bf16 At[64][32];
  __shared__ bf16 Bt[64][32];

  const int t    = threadIdx.x;
  const int lane = t & 31;
  const int wid  = t >> 5;
  const int row0 = blockIdx.x * 64;
  const int n0   = blockIdx.y * 64;

  v8f acc[4];
#pragma unroll
  for (int n = 0; n < 4; ++n) acc[n] = vzero8();

  const int am  = wid * 16 + (lane & 15);
  const int khi = (lane >> 4) * 8;
  const int kbB = (lane >> 4) * 16;

  for (int kk = 0; kk < HD; kk += 32) {
    for (int i = t; i < 64 * 32; i += 128) {
      const int r = i >> 5, c = i & 31;
      At[r][c] = Ctx[(size_t)(row0 + r) * HD + kk + c];
      Bt[r][c] = (bf16)wo[(size_t)(n0 + r) * HD + kk + c];
    }
    __syncthreads();

    v16bf a;
#pragma unroll
    for (int i = 0; i < 16; ++i) {
      const int k = (i < 8) ? (khi + i) : (16 + khi + (i - 8));
      a[i] = At[am][k];
    }
#pragma unroll
    for (int n = 0; n < 4; ++n) {
      const int bn = n * 16 + (lane & 15);
      v16bf b;
#pragma unroll
      for (int i = 0; i < 16; ++i) b[i] = Bt[bn][(i < 8) ? (kbB / 2 + i) : (16 + kbB / 2 + (i - 8))];
      acc[n] = WMMA_BF16(a, b, acc[n]);
    }
    __syncthreads();
  }

  const int mgrp = (lane >> 4) * 8;
#pragma unroll
  for (int n = 0; n < 4; ++n) {
    const int e = n0 + n * 16 + (lane & 15);
    const float bo_ = bo[e];
#pragma unroll
    for (int r = 0; r < 8; ++r) {
      const int row = row0 + wid * 16 + mgrp + r;
      out[(size_t)row * Ee + e] = acc[n][r] + bo_;
    }
  }
}

// ---------------------------------------------------------------------------
extern "C" void kernel_launch(void* const* d_in, const int* in_sizes, int n_in,
                              void* d_out, int out_size, void* d_ws, size_t ws_size,
                              hipStream_t stream) {
  const float* x  = (const float*)d_in[0];
  const float* wq = (const float*)d_in[1];
  const float* bq = (const float*)d_in[2];
  const float* wk = (const float*)d_in[3];
  const float* bk = (const float*)d_in[4];
  const float* wv = (const float*)d_in[5];
  const float* bv = (const float*)d_in[6];
  const float* wo = (const float*)d_in[7];
  const float* bo = (const float*)d_in[8];
  float* out = (float*)d_out;

  const size_t perTensorBytes = (size_t)Bb * Hh * Ss * Dk * sizeof(bf16); // 8 MB
  char* ws = (char*)d_ws;
  bf16* Qw = (bf16*)(ws);
  bf16* Kw = (bf16*)(ws + 1 * perTensorBytes);
  bf16* Vw = (bf16*)(ws + 2 * perTensorBytes);
  bf16* Cw = (bf16*)(ws + 3 * perTensorBytes);

  qkv_rope_kernel<<<dim3((Bb * Ss) / 64, 48), 128, 0, stream>>>(
      x, wq, bq, wk, bk, wv, bv, Qw, Kw, Vw);
  attention_kernel<<<dim3(Ss / 64, Bb * Hh), 128, 0, stream>>>(Qw, Kw, Vw, Cw);
  out_proj_kernel<<<dim3((Bb * Ss) / 64, Ee / 64), 128, 0, stream>>>(
      Cw, wo, bo, out);
}
